// RGLRU_55284819034482
// MI455X (gfx1250) — matrix operand order repaired and checked by reading, not compiled, not run on hardware
//
#include <hip/hip_runtime.h>

// RG-LRU for MI455X / gfx1250.
// K0: convert W_r/W_i/W_x to bf16 once (they're reused 1024x; kills in-loop cvt).
// K1: three fused bf16-WMMA GEMMs with workgroup LDS tiling of B
//     (async global->LDS staging, double buffered) + gate epilogue.
// K2: diagonal linear scan over time, one lane per (b,d) channel.

typedef __attribute__((ext_vector_type(16))) __bf16 v16bf;
typedef __attribute__((ext_vector_type(8)))  __bf16 v8bf;
typedef __attribute__((ext_vector_type(4)))  __bf16 v4bf;
typedef __attribute__((ext_vector_type(8)))  float  v8f;
typedef __attribute__((ext_vector_type(4)))  float  f32x4;
typedef __attribute__((ext_vector_type(4)))  unsigned int u32x4;
typedef __attribute__((ext_vector_type(8)))  int    i32x8;

// GCC-style vector for the async-LDS builtin (it expects __vector_size__ int4).
typedef int i32x4 __attribute__((__vector_size__(16)));
typedef __attribute__((address_space(1))) i32x4 gl_i32x4;  // global
typedef __attribute__((address_space(3))) i32x4 ld_i32x4;  // LDS

#define BSZ    4
#define TSEQ   4096
#define DMODEL 1024
#define MROWS  (BSZ * TSEQ)
#define LOG8C  2.0794415416798357f

// ---- gfx1250 async global->LDS path (guarded; sync ds staging fallback) ----
#if defined(__has_builtin)
#if __has_builtin(__builtin_amdgcn_global_load_async_to_lds_b128) && \
    __has_builtin(__builtin_amdgcn_s_wait_asynccnt)
#define HAVE_ASYNC_LDS 1
#endif
#endif
#ifndef HAVE_ASYNC_LDS
#define HAVE_ASYNC_LDS 0
#endif

// LDS B slab: double buffer x 3 matrices x 32 W-rows x (KC=64 bf16 = 128B + 16B pad).
#define KC     64
#define NCHUNK (DMODEL / KC)   // 16
#define ROWB   144             // 128B of K data + 16B pad (bank-conflict-free)
#define MATB   (32 * ROWB)     // 4608 B
#define BUFB   (3 * MATB)      // 13824 B

// Pack 16 f32 -> 16 bf16 (v_cvt_pk_bf16_f32 pairs).
static __device__ __forceinline__ v16bf cvt_pack16(f32x4 a, f32x4 b, f32x4 c, f32x4 d) {
  v16bf v;
#pragma unroll
  for (int j = 0; j < 4; ++j) {
    v[j]      = (__bf16)a[j];
    v[j + 4]  = (__bf16)b[j];
    v[j + 8]  = (__bf16)c[j];
    v[j + 12] = (__bf16)d[j];
  }
  return v;
}

// A tile (16x32 bf16, MxK): lane L = row L&15; elems 0..7 <- K h*8..+8,
// elems 8..15 <- K 16+h*8..+8  (h = L>>4).
static __device__ __forceinline__ v16bf load_a_tile(const float* p0, const float* p1) {
  const f32x4* q0 = (const f32x4*)p0;
  const f32x4* q1 = (const f32x4*)p1;
  return cvt_pack16(q0[0], q0[1], q1[0], q1[1]);
}

// B tile (32x16 bf16, KxN) from LDS: lane L = column n = L&15 (a W row),
// K 8h..+8 and 16+8h..+8 (16B each, 32B apart) -> two ds_load_b128.
static __device__ __forceinline__ v16bf lds_b_tile(const unsigned char* p) {
  const v8bf* q = (const v8bf*)p;
  v8bf lo = q[0];
  v8bf hi = q[2];
  return __builtin_shufflevector(lo, hi, 0, 1, 2, 3, 4, 5, 6, 7,
                                 8, 9, 10, 11, 12, 13, 14, 15);
}

// Scheduling fence: forces all six B tiles to be materialized (in distinct
// VGPRs) before the WMMA chain, so the 12 ds_load_b128 issue as one clause
// with a single dscnt drain instead of six exposed LDS round-trips.
static __device__ __forceinline__ void pin6(v16bf& a, v16bf& b, v16bf& c,
                                            v16bf& d, v16bf& e, v16bf& f) {
  i32x8 ta = __builtin_bit_cast(i32x8, a);
  i32x8 tb = __builtin_bit_cast(i32x8, b);
  i32x8 tc = __builtin_bit_cast(i32x8, c);
  i32x8 td = __builtin_bit_cast(i32x8, d);
  i32x8 te = __builtin_bit_cast(i32x8, e);
  i32x8 tf = __builtin_bit_cast(i32x8, f);
  asm volatile("" : "+v"(ta), "+v"(tb), "+v"(tc), "+v"(td), "+v"(te), "+v"(tf));
  a = __builtin_bit_cast(v16bf, ta);
  b = __builtin_bit_cast(v16bf, tb);
  c = __builtin_bit_cast(v16bf, tc);
  d = __builtin_bit_cast(v16bf, td);
  e = __builtin_bit_cast(v16bf, te);
  f = __builtin_bit_cast(v16bf, tf);
}

static __device__ __forceinline__ v8f wmma_bf16(v16bf a, v16bf b, v8f c) {
  return __builtin_amdgcn_wmma_f32_16x16x32_bf16(false, a, false, b, (short)0, c,
                                                 false, false);
}

static __device__ __forceinline__ float sigmoid_f(float z) {
  return 1.0f / (1.0f + __expf(-z));
}

// ---- K0: W (3 x D x D f32) -> bf16, stored contiguously [Wr|Wi|Wx] ----
__global__ __launch_bounds__(256)
void wcvt_bf16(const float* __restrict__ w0, const float* __restrict__ w1,
               const float* __restrict__ w2, __bf16* __restrict__ d) {
  const size_t nPer = (size_t)DMODEL * DMODEL;
  const size_t i = ((size_t)blockIdx.x * blockDim.x + threadIdx.x) * 4;
  const float* s;
  size_t off;
  if (i < nPer)          { s = w0; off = i; }
  else if (i < 2 * nPer) { s = w1; off = i - nPer; }
  else                   { s = w2; off = i - 2 * nPer; }
  f32x4 v = *(const f32x4*)(s + off);
  v4bf o;
  o[0] = (__bf16)v[0]; o[1] = (__bf16)v[1];
  o[2] = (__bf16)v[2]; o[3] = (__bf16)v[3];
  *(v4bf*)(d + i) = o;
}

// ---- K1: fused triple GEMM + gates ----
__global__ __launch_bounds__(256)
void rglru_gemm_gates(const float* __restrict__ x,
                      const __bf16* __restrict__ Wb,   // [3][D][D] bf16
                      const float* __restrict__ br, const float* __restrict__ bi,
                      float* __restrict__ aOut, float* __restrict__ uOut) {
  __shared__ alignas(16) unsigned char ldsB[2 * BUFB];

  const int tid    = threadIdx.x;
  const int lane   = tid & 31;
  const int wv     = tid >> 5;           // 0..7
  const int mBlk   = blockIdx.x >> 5;    // MROWS/128 = 128 row blocks
  const int nStrip = blockIdx.x & 31;    // DMODEL/32 = 32 col strips
  const int mBase  = mBlk * 128 + wv * 16;
  const int nBase  = nStrip * 32;

  const int r16  = lane & 15;
  const int half = lane >> 4;

  const float* xrow = x + (size_t)(mBase + r16) * DMODEL + half * 8;

  // Cooperative B staging: 256 threads x 16B x 3 mats per chunk.
  const int    sRow      = tid >> 3;                                   // 0..31
  const int    sSeg      = tid & 7;                                    // 0..7
  const size_t gStageOff = (size_t)(nBase + sRow) * DMODEL + sSeg * 8; // bf16 elems
  const int    lStageOff = sRow * ROWB + sSeg * 16;                    // bytes

  v8f accR0 = {}, accR1 = {}, accI0 = {}, accI1 = {}, accX0 = {}, accX1 = {};

  auto stage = [&](int kBase, int buf) {
#pragma unroll
    for (int m3 = 0; m3 < 3; ++m3) {
      const __bf16* g = Wb + (size_t)m3 * DMODEL * DMODEL + gStageOff + kBase;
      unsigned char* l = &ldsB[buf * BUFB + m3 * MATB + lStageOff];
#if HAVE_ASYNC_LDS
      __builtin_amdgcn_global_load_async_to_lds_b128(
          (gl_i32x4*)g, (ld_i32x4*)l, 0, 0);
#else
      *(u32x4*)l = *(const u32x4*)g;   // global_load_b128 + ds_store_b128
#endif
    }
  };

  stage(0, 0);

#pragma unroll 1
  for (int c = 0; c < NCHUNK; ++c) {
    if (c + 1 < NCHUNK) stage((c + 1) * KC, (c + 1) & 1);
#if HAVE_ASYNC_LDS
    if (c + 1 < NCHUNK) __builtin_amdgcn_s_wait_asynccnt(3);  // chunk c landed
    else                __builtin_amdgcn_s_wait_asynccnt(0);
#endif
    __syncthreads();

    const unsigned char* bBase = &ldsB[(c & 1) * BUFB];
    const int kBase = c * KC;

    // Issue both A-tile global loads early (cvt lands after the waits).
    v16bf A0 = load_a_tile(xrow + kBase,      xrow + kBase + 16);
    v16bf A1 = load_a_tile(xrow + kBase + 32, xrow + kBase + 48);

    const unsigned char* p0 = bBase + r16 * ROWB + half * 16;         // cols 0..15
    const unsigned char* p1 = bBase + (16 + r16) * ROWB + half * 16;  // cols 16..31

    // ---- K-step 0: batch all 12 ds_load_b128, then 6 WMMAs ----
    {
      v16bf bR0 = lds_b_tile(p0 + 0 * MATB);
      v16bf bR1 = lds_b_tile(p1 + 0 * MATB);
      v16bf bI0 = lds_b_tile(p0 + 1 * MATB);
      v16bf bI1 = lds_b_tile(p1 + 1 * MATB);
      v16bf bX0 = lds_b_tile(p0 + 2 * MATB);
      v16bf bX1 = lds_b_tile(p1 + 2 * MATB);
      pin6(bR0, bR1, bI0, bI1, bX0, bX1);
      accR0 = wmma_bf16(A0, bR0, accR0);
      accR1 = wmma_bf16(A0, bR1, accR1);
      accI0 = wmma_bf16(A0, bI0, accI0);
      accI1 = wmma_bf16(A0, bI1, accI1);
      accX0 = wmma_bf16(A0, bX0, accX0);
      accX1 = wmma_bf16(A0, bX1, accX1);
    }
    // ---- K-step 1 (next 64B along K in the LDS rows) ----
    {
      v16bf bR0 = lds_b_tile(p0 + 0 * MATB + 64);
      v16bf bR1 = lds_b_tile(p1 + 0 * MATB + 64);
      v16bf bI0 = lds_b_tile(p0 + 1 * MATB + 64);
      v16bf bI1 = lds_b_tile(p1 + 1 * MATB + 64);
      v16bf bX0 = lds_b_tile(p0 + 2 * MATB + 64);
      v16bf bX1 = lds_b_tile(p1 + 2 * MATB + 64);
      pin6(bR0, bR1, bI0, bI1, bX0, bX1);
      accR0 = wmma_bf16(A1, bR0, accR0);
      accR1 = wmma_bf16(A1, bR1, accR1);
      accI0 = wmma_bf16(A1, bI0, accI0);
      accI1 = wmma_bf16(A1, bI1, accI1);
      accX0 = wmma_bf16(A1, bX0, accX0);
      accX1 = wmma_bf16(A1, bX1, accX1);
    }
    __syncthreads();   // all waves done reading before this buf is re-staged
  }

  // Epilogue: C/D layout -> lane holds N = lane&15, M = mBase + (lane>>4)*8 + r.
  const int   n0   = nBase + r16;
  const int   n1   = n0 + 16;
  const float vbr0 = br[n0], vbr1 = br[n1];
  const float vbi0 = bi[n0], vbi1 = bi[n1];
  const int   mRow = mBase + half * 8;

#pragma unroll
  for (int r = 0; r < 8; ++r) {
    const size_t row = (size_t)(mRow + r) * DMODEL;
    {
      float rg = sigmoid_f(accR0[r] + vbr0);
      float ig = sigmoid_f(accI0[r] + vbi0);
      float av = sigmoid_f(-LOG8C * rg);       // exp(-softplus(z)) == sigmoid(-z)
      float gt = __fsqrt_rn(fmaxf(1.0f - av * av, 1e-6f));
      aOut[row + n0] = av;
      uOut[row + n0] = gt * ig * accX0[r];
    }
    {
      float rg = sigmoid_f(accR1[r] + vbr1);
      float ig = sigmoid_f(accI1[r] + vbi1);
      float av = sigmoid_f(-LOG8C * rg);
      float gt = __fsqrt_rn(fmaxf(1.0f - av * av, 1e-6f));
      aOut[row + n1] = av;
      uOut[row + n1] = gt * ig * accX1[r];
    }
  }
}

// ---- K2: h_t = a_t * h_{t-1} + u_t, seq[] holds a and is overwritten with h ----
__global__ __launch_bounds__(256)
void rglru_scan(const float* __restrict__ h0,
                const float* __restrict__ uBuf,
                float* __restrict__ seq,
                float* __restrict__ hLast) {
  const int c = blockIdx.x * blockDim.x + threadIdx.x;  // channel in [0, BSZ*DMODEL)
  const int b = c >> 10;
  const int d = c & (DMODEL - 1);
  float h = h0[c];
  const size_t base = ((size_t)b * TSEQ) * DMODEL + d;

  for (int t = 0; t < TSEQ; t += 4) {
    const size_t idx = base + (size_t)t * DMODEL;
    __builtin_prefetch(seq  + idx + 16 * DMODEL, 0, 1);   // global_prefetch_b8
    __builtin_prefetch(uBuf + idx + 16 * DMODEL, 0, 1);
    float a0 = seq[idx];
    float u0 = uBuf[idx];
    float a1 = seq[idx + 1 * DMODEL];
    float u1 = uBuf[idx + 1 * DMODEL];
    float a2 = seq[idx + 2 * DMODEL];
    float u2 = uBuf[idx + 2 * DMODEL];
    float a3 = seq[idx + 3 * DMODEL];
    float u3 = uBuf[idx + 3 * DMODEL];
    h = __builtin_fmaf(a0, h, u0); seq[idx]              = h;
    h = __builtin_fmaf(a1, h, u1); seq[idx + 1 * DMODEL] = h;
    h = __builtin_fmaf(a2, h, u2); seq[idx + 2 * DMODEL] = h;
    h = __builtin_fmaf(a3, h, u3); seq[idx + 3 * DMODEL] = h;
  }
  hLast[c] = h;
}

extern "C" void kernel_launch(void* const* d_in, const int* in_sizes, int n_in,
                              void* d_out, int out_size, void* d_ws, size_t ws_size,
                              hipStream_t stream) {
  (void)in_sizes; (void)n_in; (void)out_size; (void)ws_size;
  const float* x  = (const float*)d_in[0];
  const float* h0 = (const float*)d_in[1];
  const float* Wr = (const float*)d_in[2];
  const float* br = (const float*)d_in[3];
  const float* Wi = (const float*)d_in[4];
  const float* bi = (const float*)d_in[5];
  const float* Wx = (const float*)d_in[6];

  float*  seq   = (float*)d_out;                       // B*T*D sequence output
  float*  hLast = seq + (size_t)MROWS * DMODEL;        // + B*D final state
  float*  uBuf  = (float*)d_ws;                        // B*T*D f32
  __bf16* Wb    = (__bf16*)((char*)d_ws +
                            (size_t)MROWS * DMODEL * sizeof(float)); // 3*D*D bf16

  // K0: weight conversion (3*D*D/4 elems per thread-group-of-4).
  wcvt_bf16<<<dim3(3 * DMODEL * DMODEL / 4 / 256), dim3(256), 0, stream>>>(
      Wr, Wi, Wx, Wb);

  // K1: 128-row x 32-col blocks, 8 waves each.
  rglru_gemm_gates<<<dim3((MROWS / 128) * (DMODEL / 32)), dim3(256), 0, stream>>>(
      x, Wb, br, bi, seq, uBuf);

  // K2: serial scan, one lane per channel.
  rglru_scan<<<dim3((BSZ * DMODEL) / 256), dim3(256), 0, stream>>>(h0, uBuf, seq,
                                                                   hLast);
}
